// Model3_28733331210263
// MI455X (gfx1250) — matrix operand order repaired and checked by reading, not compiled, not run on hardware
//
#include <hip/hip_runtime.h>
#include <hip/hip_bf16.h>

// Problem sizes (from reference)
#define TT 512
#define BB 128
#define PP 10
#define II 50
#define HH 512
#define NC 8
#define AK 576           // padded K: [h(512) | emb(50) | zeros(14)]
#define NKC 18           // AK/32 k-chunks
#define NWG 32           // persistent workgroups
#define THREADS 256      // 8 waves per WG -> 256 waves total

// Workspace byte offsets (all 16B aligned)
#define OFF_BAR 0u
#define OFF_U   256u                      // u: 128 x 64 f32         (32768 B)
#define OFF_A   33024u                    // a_bf16: 128 x 576 bf16  (147456 B)
#define OFF_H   180480u                   // h: 128 x 512 f32        (262144 B)
#define OFF_C   442624u                   // c: 128 x 512 f32        (262144 B)
#define OFF_WA  704768u                   // Wa swizzled: 64 blk x 512 bf16 (65536 B)
#define OFF_WC  770304u                   // Wc swizzled: 2304 blk x 512 bf16 (2359296 B)
#define ZERO_WORDS (OFF_WA / 4u)          // zero everything up to the weights

typedef __attribute__((ext_vector_type(16))) __bf16 v16bf;
typedef __attribute__((ext_vector_type(8)))  __bf16 v8bf;
typedef __attribute__((ext_vector_type(8)))  float  v8f;

__device__ __forceinline__ __bf16 f2bf(float f) {
    unsigned u = __float_as_uint(f);
    unsigned r = (u + 0x7FFFu + ((u >> 16) & 1u)) >> 16;   // RNE
    unsigned short s = (unsigned short)r;
    __bf16 b;
    __builtin_memcpy(&b, &s, 2);
    return b;
}

__device__ __forceinline__ v16bf load_frag(const __bf16* p0, const __bf16* p1) {
    v8bf lo = *(const v8bf*)p0;
    v8bf hi = *(const v8bf*)p1;
    v16bf r;
#pragma unroll
    for (int e = 0; e < 8; ++e) { r[e] = lo[e]; r[e + 8] = hi[e]; }
    return r;
}

// Grid barrier, agent (device) scope only — no SYS-scope traffic.
__device__ __forceinline__ void gsync(unsigned* cnt, unsigned* gen) {
    __threadfence();
    __syncthreads();
    if (threadIdx.x == 0) {
        unsigned g = __hip_atomic_load(gen, __ATOMIC_RELAXED, __HIP_MEMORY_SCOPE_AGENT);
        if (__hip_atomic_fetch_add(cnt, 1u, __ATOMIC_ACQ_REL, __HIP_MEMORY_SCOPE_AGENT)
                == NWG - 1u) {
            __hip_atomic_store(cnt, 0u, __ATOMIC_RELAXED, __HIP_MEMORY_SCOPE_AGENT);
            __hip_atomic_store(gen, g + 1u, __ATOMIC_RELEASE, __HIP_MEMORY_SCOPE_AGENT);
        } else {
            while (__hip_atomic_load(gen, __ATOMIC_ACQUIRE, __HIP_MEMORY_SCOPE_AGENT) == g)
                __builtin_amdgcn_s_sleep(1);
        }
    }
    __syncthreads();
    __threadfence();
}

// One-time: zero state, build bf16 fragment-swizzled weights.
// Wa_swz: blocks (nt,kc) nt<4,kc<16; lane l: N=nt*16+(l&15), K=kc*32+e+16*(l>>4);
//         value = Wa[K,N] for N<50, ba[K] for N==50 (folds h.ba into scores), else 0.
// Wc_swz: blocks (jt',kc) jt'<128,kc<18; j=jt'*16+(l&15), K as above;
//         value = W_hh[j,K] for K<512, W_ih[j,K-512] for K<562, else 0.
__global__ void lstm_init_kernel(const float* __restrict__ Wa, const float* __restrict__ ba,
                                 const float* __restrict__ W_ih, const float* __restrict__ W_hh,
                                 unsigned char* __restrict__ ws) {
    size_t stride = (size_t)gridDim.x * blockDim.x;
    size_t tid = (size_t)blockIdx.x * blockDim.x + threadIdx.x;

    unsigned* z = (unsigned*)ws;
    for (size_t i = tid; i < ZERO_WORDS; i += stride) z[i] = 0u;

    __bf16* wa = (__bf16*)(ws + OFF_WA);
    for (size_t idx = tid; idx < (size_t)64 * 512; idx += stride) {
        int blk = (int)(idx >> 9), rem = (int)(idx & 511);
        int lane = rem >> 4, e = rem & 15;
        int nt = blk >> 4, kc = blk & 15;
        int N = nt * 16 + (lane & 15), half = lane >> 4;
        int K = kc * 32 + e + 16 * half;
        float v = 0.f;
        if (N < II)       v = Wa[K * II + N];
        else if (N == II) v = ba[K];
        wa[idx] = f2bf(v);
    }

    __bf16* wc = (__bf16*)(ws + OFF_WC);
    for (size_t idx = tid; idx < (size_t)2304 * 512; idx += stride) {
        int blk = (int)(idx >> 9), rem = (int)(idx & 511);
        int lane = rem >> 4, e = rem & 15;
        int jt = blk / NKC, kc = blk % NKC;
        int j = jt * 16 + (lane & 15), half = lane >> 4;
        int K = kc * 32 + e + 16 * half;
        float v = 0.f;
        if (K < HH)            v = W_hh[(size_t)j * HH + K];
        else if (K < HH + II)  v = W_ih[j * II + (K - HH)];
        wc[idx] = f2bf(v);
    }
}

__global__ void __launch_bounds__(THREADS, 1)
lstm_persistent_kernel(const float* __restrict__ x, const float* __restrict__ mask,
                       const float* __restrict__ b_ih, const float* __restrict__ b_hh,
                       const float* __restrict__ W_fc, const float* __restrict__ b_fc,
                       unsigned char* __restrict__ ws, float* __restrict__ out) {
    unsigned* cnt = (unsigned*)(ws + OFF_BAR);
    unsigned* gen = cnt + 1;
    float*  u    = (float*) (ws + OFF_U);
    __bf16* a_bf = (__bf16*)(ws + OFF_A);
    float*  h_ws = (float*) (ws + OFF_H);
    float*  c_ws = (float*) (ws + OFF_C);
    const __bf16* wa_swz = (const __bf16*)(ws + OFF_WA);
    const __bf16* wc_swz = (const __bf16*)(ws + OFF_WC);

    const int tid  = blockIdx.x * blockDim.x + threadIdx.x;
    const int wid  = tid >> 5;
    const int lane = threadIdx.x & 31;
    const int n16  = lane & 15;
    const int half = lane >> 4;

    // ensure init kernel's writes are visible
    gsync(cnt, gen);

    for (int t = 0; t < TT; ++t) {
        // ---------------- Phase A: u = h_bf16 @ [Wa|ba]  (128x64, K=512, WMMA) ----
        if (wid < 32) {
            int rt = wid >> 2, nt = wid & 3;
            v8f acc = {};
            const __bf16* abase = a_bf + (size_t)(rt * 16 + n16) * AK;
#pragma unroll 4
            for (int kc = 0; kc < 16; ++kc) {
                const __bf16* ap = abase + kc * 32 + 16 * half;
                v16bf A = load_frag(ap, ap + 8);
                const __bf16* wp = wa_swz + (((size_t)(nt * 16 + kc) * 32 + lane) << 4);
                v16bf Bf = load_frag(wp, wp + 8);
                acc = __builtin_amdgcn_wmma_f32_16x16x32_bf16(
                    false, A, false, Bf, (short)0, acc, false, false);
            }
#pragma unroll
            for (int r = 0; r < 8; ++r) {
                int b = rt * 16 + r + 8 * half;
                u[b * 64 + nt * 16 + n16] = acc[r];
            }
        }
        gsync(cnt, gen);

        // ---------------- Phase B: scores -> softmax -> emb (one wave per batch) --
        if (wid < BB) {
            int b = wid;
            const float* urow = u + b * 64;
            float u0 = (lane < II) ? urow[lane] : 0.f;
            float u1 = (lane + 32 < II) ? urow[lane + 32] : 0.f;
            float hb = urow[II];   // folded h.ba term
            const float* xrow = x + ((size_t)(b * TT + t)) * (PP * II);
            const float* mrow = mask + ((size_t)(b * TT + t)) * (PP * II);
            // prefetch next timestep's pose slice (2000 B, 32 lanes x 64 B strides)
            if (t + 1 < TT)
                __builtin_prefetch(xrow + (PP * II) + lane * 16, 0, 1);
            float sc[PP];
#pragma unroll
            for (int p = 0; p < PP; ++p) {
                float xa = (lane < II) ? xrow[p * II + lane] : 0.f;
                float xb = (lane + 32 < II) ? xrow[p * II + lane + 32] : 0.f;
                float part = u0 * xa + u1 * xb;
#pragma unroll
                for (int o = 16; o; o >>= 1) part += __shfl_xor(part, o, 32);
                sc[p] = (mrow[p * II] > 0.f) ? (part + hb) : -1e9f;
            }
            float mx = sc[0];
#pragma unroll
            for (int p = 1; p < PP; ++p) mx = fmaxf(mx, sc[p]);
            float se = 0.f, w[PP];
#pragma unroll
            for (int p = 0; p < PP; ++p) { w[p] = __expf(sc[p] - mx); se += w[p]; }
            float inv = 1.f / se;
            float e0 = 0.f, e1 = 0.f;
#pragma unroll
            for (int p = 0; p < PP; ++p) {
                float wp = w[p] * inv;
                if (lane < II)      e0 += wp * xrow[p * II + lane];
                if (lane + 32 < II) e1 += wp * xrow[p * II + lane + 32];
            }
            __bf16* arow = a_bf + (size_t)b * AK + HH;
            if (lane < II)      arow[lane]      = f2bf(e0);
            if (lane + 32 < II) arow[lane + 32] = f2bf(e1);
        }
        gsync(cnt, gen);

        // ---------------- Phase C: gates = [h|emb] @ Wc^T + bias; LSTM cell -------
        {
            int rt = wid >> 5, jt = wid & 31;   // 8 row-tiles x 32 col-tile groups
            v8f acc0 = {}, acc1 = {}, acc2 = {}, acc3 = {};
            const __bf16* abase = a_bf + (size_t)(rt * 16 + n16) * AK;
#pragma unroll 2
            for (int kc = 0; kc < NKC; ++kc) {
                const __bf16* ap = abase + kc * 32 + 16 * half;
                v16bf A = load_frag(ap, ap + 8);
                const __bf16* w0 = wc_swz + (((size_t)((0 * 32 + jt) * NKC + kc) * 32 + lane) << 4);
                const __bf16* w1 = wc_swz + (((size_t)((1 * 32 + jt) * NKC + kc) * 32 + lane) << 4);
                const __bf16* w2 = wc_swz + (((size_t)((2 * 32 + jt) * NKC + kc) * 32 + lane) << 4);
                const __bf16* w3 = wc_swz + (((size_t)((3 * 32 + jt) * NKC + kc) * 32 + lane) << 4);
                v16bf B0 = load_frag(w0, w0 + 8);
                acc0 = __builtin_amdgcn_wmma_f32_16x16x32_bf16(false, A, false, B0, (short)0, acc0, false, false);
                v16bf B1 = load_frag(w1, w1 + 8);
                acc1 = __builtin_amdgcn_wmma_f32_16x16x32_bf16(false, A, false, B1, (short)0, acc1, false, false);
                v16bf B2 = load_frag(w2, w2 + 8);
                acc2 = __builtin_amdgcn_wmma_f32_16x16x32_bf16(false, A, false, B2, (short)0, acc2, false, false);
                v16bf B3 = load_frag(w3, w3 + 8);
                acc3 = __builtin_amdgcn_wmma_f32_16x16x32_bf16(false, A, false, B3, (short)0, acc3, false, false);
            }
            int j = jt * 16 + n16;
            float bi = b_ih[j]            + b_hh[j];
            float bf = b_ih[HH + j]       + b_hh[HH + j];
            float bg = b_ih[2 * HH + j]   + b_hh[2 * HH + j];
            float bo = b_ih[3 * HH + j]   + b_hh[3 * HH + j];
#pragma unroll
            for (int r = 0; r < 8; ++r) {
                int b = rt * 16 + r + 8 * half;
                float ig = acc0[r] + bi;
                float fg = acc1[r] + bf;
                float gg = acc2[r] + bg;
                float og = acc3[r] + bo;
                float si = 1.f / (1.f + __expf(-ig));
                float sf = 1.f / (1.f + __expf(-fg));
                float so = 1.f / (1.f + __expf(-og));
                float tg = tanhf(gg);
                size_t idx = (size_t)b * HH + j;
                float cn = sf * c_ws[idx] + si * tg;
                float hn = so * tanhf(cn);
                c_ws[idx] = cn;
                h_ws[idx] = hn;
                a_bf[(size_t)b * AK + j] = f2bf(hn);
            }
        }
        gsync(cnt, gen);
    }

    // ---------------- Final classifier: out = h @ W_fc^T + b_fc (128x8) ----------
    if (tid < BB * NC) {
        int b = tid >> 3, cc = tid & 7;
        const float* hrow = h_ws + (size_t)b * HH;
        const float* wrow = W_fc + cc * HH;
        float s = b_fc[cc];
#pragma unroll 8
        for (int k = 0; k < HH; ++k) s += hrow[k] * wrow[k];
        out[tid] = s;
    }
}

extern "C" void kernel_launch(void* const* d_in, const int* in_sizes, int n_in,
                              void* d_out, int out_size, void* d_ws, size_t ws_size,
                              hipStream_t stream) {
    const float* x    = (const float*)d_in[0];
    const float* mask = (const float*)d_in[1];
    const float* Wa   = (const float*)d_in[2];
    const float* ba   = (const float*)d_in[3];
    const float* W_ih = (const float*)d_in[4];
    const float* W_hh = (const float*)d_in[5];
    const float* b_ih = (const float*)d_in[6];
    const float* b_hh = (const float*)d_in[7];
    const float* W_fc = (const float*)d_in[8];
    const float* b_fc = (const float*)d_in[9];
    float* out = (float*)d_out;
    unsigned char* ws = (unsigned char*)d_ws;

    hipLaunchKernelGGL(lstm_init_kernel, dim3(1024), dim3(THREADS), 0, stream,
                       Wa, ba, W_ih, W_hh, ws);
    hipLaunchKernelGGL(lstm_persistent_kernel, dim3(NWG), dim3(THREADS), 0, stream,
                       x, mask, b_ih, b_hh, W_fc, b_fc, ws, out);
}
